// SASRecModel_24129126269360
// MI455X (gfx1250) — matrix operand order repaired and checked by reading, not compiled, not run on hardware
//
#include <hip/hip_runtime.h>
#include <hip/hip_bf16.h>
#include <math.h>

// ---------------------------------------------------------------------------
// Multimodal MoE fusion (B=64, L=128, H=1024, E=4, k=2) for gfx1250 / MI455X.
// GEMMs on V_WMMA_F32_16X16X32_BF16; tile staging via GLOBAL_LOAD_ASYNC_TO_LDS
// (ASYNCcnt path) with double-buffered LDS so staging overlaps WMMA compute.
// ---------------------------------------------------------------------------

typedef __attribute__((ext_vector_type(16))) __bf16 v16bf;
typedef __attribute__((ext_vector_type(8)))  float  v8f;
typedef __attribute__((ext_vector_type(4)))  int    v4i;

#define HDIM   1024
#define BLTOK  8192      // B*L tokens
#define NEXP   4

constexpr int BM  = 32;   // block tile M (2 wave-rows of 16)
constexpr int BN  = 128;  // block tile N (4 wave-cols x 2 sub-tiles of 16)
constexpr int BK  = 32;   // K step == WMMA K
constexpr int PAD = 8;    // bf16 pad: keeps 16B alignment, row stride 80B
constexpr int BKP = BK + PAD;

#if __has_builtin(__builtin_amdgcn_global_load_async_to_lds_b128)
#define USE_ASYNC_LDS 1
// Builtin signature (per hipcc diagnostic): (v4i AS1*, v4i AS3*, imm, imm)
typedef __attribute__((address_space(1))) v4i* gv4i_p;
typedef __attribute__((address_space(3))) v4i* sv4i_p;
#endif

__device__ inline void wait_async() {
#if defined(USE_ASYNC_LDS)
#if __has_builtin(__builtin_amdgcn_s_wait_asynccnt)
  __builtin_amdgcn_s_wait_asynccnt(0);
#else
  asm volatile("s_wait_asynccnt 0" ::: "memory");
#endif
#endif
}

// Stage `rows` x 32 bf16 tile from global (row stride `stride` elems) to LDS.
__device__ inline void stage_rows(const __bf16* __restrict__ src, size_t stride,
                                  __bf16 (*__restrict__ dst)[BKP], int rows,
                                  int tid) {
#if defined(USE_ASYNC_LDS)
  // 16B segments: 4 per 32-elem row; LDS row stride 80B keeps 16B alignment.
  for (int i = tid; i < rows * 4; i += 256) {
    const int r = i >> 2, seg = i & 3;
    __builtin_amdgcn_global_load_async_to_lds_b128(
        (gv4i_p)(src + (size_t)r * stride + seg * 8),
        (sv4i_p)&dst[r][seg * 8],
        0, 0);
  }
#else
  for (int i = tid; i < rows * 32; i += 256) {
    const int r = i >> 5, c = i & 31;
    dst[r][c] = src[(size_t)r * stride + c];
  }
#endif
}

// A fragment (16x32 bf16): lane M = l&15; elem e -> K=(e<8?e:e+8)+8*(l>=16).
__device__ inline v16bf frag_a(const __bf16 t[][BKP], int rowBase, int lane) {
  const int r = rowBase + (lane & 15);
  const int off = (lane >> 4) * 8;
  v16bf a;
#pragma unroll
  for (int p = 0; p < 8; ++p) {
    const int kb = (p < 4 ? 2 * p : 2 * p + 8) + off;
    a[2 * p]     = t[r][kb];
    a[2 * p + 1] = t[r][kb + 1];
  }
  return a;
}

// B fragment (32x16 bf16): lane N = l&15; elem e -> K=(e<8?e:e+8)+8*(l>=16).
__device__ inline v16bf frag_b(const __bf16 t[][BKP], int colBase, int lane) {
  const int n = colBase + (lane & 15);
  const int off = (lane >> 4) * 8;
  v16bf b;
#pragma unroll
  for (int e = 0; e < 16; ++e) b[e] = t[n][off + (e < 8 ? e : e + 8)];
  return b;
}

#define WMMA_BF16(A, B, C) \
  __builtin_amdgcn_wmma_f32_16x16x32_bf16(false, (A), false, (B), (short)0, (C), false, false)

// ---------------------------------------------------------------------------
// Kernel 0: fp32 -> bf16 conversion (weights & activations), 4 elems/thread.
// ---------------------------------------------------------------------------
__global__ __launch_bounds__(256) void cvt_bf16_kernel(
    const float* __restrict__ src, __bf16* __restrict__ dst) {
  const size_t i = ((size_t)blockIdx.x * 256 + threadIdx.x) * 4;
  const float4 v = *(const float4*)(src + i);
  dst[i + 0] = (__bf16)v.x;
  dst[i + 1] = (__bf16)v.y;
  dst[i + 2] = (__bf16)v.z;
  dst[i + 3] = (__bf16)v.w;
}

// ---------------------------------------------------------------------------
// Kernel 1: z = (X @ Wmu^T + bmu) + exp(X @ Wsig^T + bsig) * eps   (bf16 out)
// One A tile feeds 4 WMMAs (2 N sub-tiles x {mu, sig}); double-buffered LDS.
// ---------------------------------------------------------------------------
__global__ __launch_bounds__(256) void proj_z_kernel(
    const __bf16* __restrict__ X,
    const __bf16* __restrict__ Wmu,  const float* __restrict__ bmu,
    const __bf16* __restrict__ Wsig, const float* __restrict__ bsig,
    const float* __restrict__ eps,
    __bf16* __restrict__ zout) {
  __shared__ __bf16 sA [2][BM][BKP];
  __shared__ __bf16 sBm[2][BN][BKP];
  __shared__ __bf16 sBs[2][BN][BKP];

  const int tid  = threadIdx.x;
  const int lane = tid & 31;
  const int wave = tid >> 5;
  const int wm = wave >> 2, wn = wave & 3;   // 2 x 4 wave grid
  const int n0 = blockIdx.x * BN;
  const int m0 = blockIdx.y * BM;
  constexpr int NK = HDIM / BK;

  auto stage = [&](int buf, int k0) {
    stage_rows(X    + (size_t)m0 * HDIM + k0, HDIM, sA[buf],  BM, tid);
    stage_rows(Wmu  + (size_t)n0 * HDIM + k0, HDIM, sBm[buf], BN, tid);
    stage_rows(Wsig + (size_t)n0 * HDIM + k0, HDIM, sBs[buf], BN, tid);
  };

  v8f accM0 = {}, accM1 = {}, accS0 = {}, accS1 = {};

  stage(0, 0);
  wait_async();
  __syncthreads();

  for (int kt = 0; kt < NK; ++kt) {
    const int cur = kt & 1;
    if (kt + 1 < NK) stage(cur ^ 1, (kt + 1) * BK);

    const v16bf a   = frag_a(sA[cur], wm * 16, lane);
    const v16bf bm0 = frag_b(sBm[cur], wn * 32,      lane);
    const v16bf bm1 = frag_b(sBm[cur], wn * 32 + 16, lane);
    const v16bf bs0 = frag_b(sBs[cur], wn * 32,      lane);
    const v16bf bs1 = frag_b(sBs[cur], wn * 32 + 16, lane);
    accM0 = WMMA_BF16(a, bm0, accM0);
    accM1 = WMMA_BF16(a, bm1, accM1);
    accS0 = WMMA_BF16(a, bs0, accS0);
    accS1 = WMMA_BF16(a, bs1, accS1);

    wait_async();
    __syncthreads();
  }

  // C layout: lane l, VGPR r -> M = r + 8*(l>=16), N = l&15
  const int l16 = lane & 15, lhi = lane >> 4;
  const int row0 = m0 + wm * 16;
#pragma unroll
  for (int t = 0; t < 2; ++t) {
    const v8f& aM = t ? accM1 : accM0;
    const v8f& aS = t ? accS1 : accS0;
    const int n = n0 + wn * 32 + t * 16 + l16;
#pragma unroll
    for (int r = 0; r < 8; ++r) {
      const int m = row0 + r + lhi * 8;
      const float mu = aM[r] + bmu[n];
      const float sg = __expf(aS[r] + bsig[n]);
      zout[(size_t)m * HDIM + n] = (__bf16)(mu + sg * eps[(size_t)m * HDIM + n]);
    }
  }
}

// ---------------------------------------------------------------------------
// Kernel 2: softmax -> top-2 -> scatter -> renormalize.  One wave per token.
// ---------------------------------------------------------------------------
__global__ __launch_bounds__(256) void gate_kernel(
    const __bf16* __restrict__ z, const float* __restrict__ Wg,
    const float* __restrict__ bg, float* __restrict__ gate) {
  const int tid  = threadIdx.x;
  const int lane = tid & 31;
  const int tok  = blockIdx.x * 8 + (tid >> 5);

  float a0 = 0.f, a1 = 0.f, a2 = 0.f, a3 = 0.f;
  for (int h = lane; h < HDIM; h += 32) {
    const float zv = (float)z[(size_t)tok * HDIM + h];
    a0 += zv * Wg[h];
    a1 += zv * Wg[HDIM + h];
    a2 += zv * Wg[2 * HDIM + h];
    a3 += zv * Wg[3 * HDIM + h];
  }
#pragma unroll
  for (int off = 16; off > 0; off >>= 1) {
    a0 += __shfl_down(a0, off, 32);
    a1 += __shfl_down(a1, off, 32);
    a2 += __shfl_down(a2, off, 32);
    a3 += __shfl_down(a3, off, 32);
  }
  if (lane == 0) {
    float lg[4] = {a0 + bg[0], a1 + bg[1], a2 + bg[2], a3 + bg[3]};
    const float mx = fmaxf(fmaxf(lg[0], lg[1]), fmaxf(lg[2], lg[3]));
    float p[4], s = 0.f;
#pragma unroll
    for (int e = 0; e < 4; ++e) { p[e] = __expf(lg[e] - mx); s += p[e]; }
#pragma unroll
    for (int e = 0; e < 4; ++e) p[e] /= s;
    int i1 = 0;
    for (int e = 1; e < 4; ++e) if (p[e] > p[i1]) i1 = e;
    int i2 = (i1 == 0) ? 1 : 0;
    for (int e = 0; e < 4; ++e) if (e != i1 && p[e] > p[i2]) i2 = e;
    const float s2 = p[i1] + p[i2];
#pragma unroll
    for (int e = 0; e < 4; ++e)
      gate[(size_t)tok * NEXP + e] = (e == i1 || e == i2) ? p[e] / s2 : 0.f;
  }
}

// ---------------------------------------------------------------------------
// Kernel 3: dense MoE: out = sum_e gate[.,e] * (z @ We[e]^T + be[e])
// One A tile feeds 8 WMMAs (4 experts x 2 N sub-tiles); double-buffered LDS.
// ---------------------------------------------------------------------------
__global__ __launch_bounds__(256) void moe_kernel(
    const __bf16* __restrict__ z,
    const __bf16* __restrict__ We,  // [E,H,H] bf16
    const float* __restrict__ be,   // [E,H]
    const float* __restrict__ gate, // [BL,E]
    __bf16* __restrict__ out) {
  __shared__ __bf16 sA[2][BM][BKP];
  __shared__ __bf16 sB[2][NEXP][BN][BKP];
  __shared__ float  sG[BM][NEXP];

  const int tid  = threadIdx.x;
  const int lane = tid & 31;
  const int wave = tid >> 5;
  const int wm = wave >> 2, wn = wave & 3;
  const int n0 = blockIdx.x * BN;
  const int m0 = blockIdx.y * BM;
  constexpr int NK = HDIM / BK;

  for (int i = tid; i < BM * NEXP; i += 256)
    sG[i >> 2][i & 3] = gate[(size_t)(m0 + (i >> 2)) * NEXP + (i & 3)];

  auto stage = [&](int buf, int k0) {
    stage_rows(z + (size_t)m0 * HDIM + k0, HDIM, sA[buf], BM, tid);
#pragma unroll
    for (int e = 0; e < NEXP; ++e)
      stage_rows(We + ((size_t)e * HDIM + n0) * HDIM + k0, HDIM, sB[buf][e],
                 BN, tid);
  };

  const v8f zacc = {};
  v8f acc[NEXP][2];
#pragma unroll
  for (int e = 0; e < NEXP; ++e) { acc[e][0] = zacc; acc[e][1] = zacc; }

  stage(0, 0);
  wait_async();
  __syncthreads();

  for (int kt = 0; kt < NK; ++kt) {
    const int cur = kt & 1;
    if (kt + 1 < NK) stage(cur ^ 1, (kt + 1) * BK);

    // Hoist all fragment loads so the 8 WMMAs can issue back-to-back.
    const v16bf a = frag_a(sA[cur], wm * 16, lane);
    v16bf bf[NEXP][2];
#pragma unroll
    for (int e = 0; e < NEXP; ++e) {
      bf[e][0] = frag_b(sB[cur][e], wn * 32,      lane);
      bf[e][1] = frag_b(sB[cur][e], wn * 32 + 16, lane);
    }
#pragma unroll
    for (int e = 0; e < NEXP; ++e) {
      acc[e][0] = WMMA_BF16(a, bf[e][0], acc[e][0]);
      acc[e][1] = WMMA_BF16(a, bf[e][1], acc[e][1]);
    }

    wait_async();
    __syncthreads();
  }

  const int l16 = lane & 15, lhi = lane >> 4;
#pragma unroll
  for (int t = 0; t < 2; ++t) {
    const int n = n0 + wn * 32 + t * 16 + l16;
#pragma unroll
    for (int r = 0; r < 8; ++r) {
      const int mLoc = wm * 16 + r + lhi * 8;
      float o = 0.f;
#pragma unroll
      for (int e = 0; e < NEXP; ++e)
        o += sG[mLoc][e] * (acc[e][t][r] + be[e * HDIM + n]);
      out[(size_t)(m0 + mLoc) * HDIM + n] = (__bf16)o;
    }
  }
}

// ---------------------------------------------------------------------------
// Kernel 4: h = concat(t_out, i_out) @ W_fuse^T + b_fuse   (K = 2048, f32 out)
// ---------------------------------------------------------------------------
__global__ __launch_bounds__(256) void fuse_kernel(
    const __bf16* __restrict__ tA, const __bf16* __restrict__ iA,
    const __bf16* __restrict__ Wf,  // [H, 2H] bf16
    const float* __restrict__ bfv,  // [H]
    float* __restrict__ h) {
  __shared__ __bf16 sA[2][BM][BKP];
  __shared__ __bf16 sB[2][BN][BKP];

  const int tid  = threadIdx.x;
  const int lane = tid & 31;
  const int wave = tid >> 5;
  const int wm = wave >> 2, wn = wave & 3;
  const int n0 = blockIdx.x * BN;
  const int m0 = blockIdx.y * BM;
  constexpr int NK = (2 * HDIM) / BK;

  auto stage = [&](int buf, int k0) {
    const __bf16* Asrc = (k0 < HDIM) ? tA : iA;  // 32-wide tile in one half
    const int kb = k0 & (HDIM - 1);
    stage_rows(Asrc + (size_t)m0 * HDIM + kb, HDIM, sA[buf], BM, tid);
    stage_rows(Wf + (size_t)n0 * (2 * HDIM) + k0, 2 * HDIM, sB[buf], BN, tid);
  };

  v8f acc0 = {}, acc1 = {};

  stage(0, 0);
  wait_async();
  __syncthreads();

  for (int kt = 0; kt < NK; ++kt) {
    const int cur = kt & 1;
    if (kt + 1 < NK) stage(cur ^ 1, (kt + 1) * BK);

    const v16bf a  = frag_a(sA[cur], wm * 16, lane);
    const v16bf b0 = frag_b(sB[cur], wn * 32,      lane);
    const v16bf b1 = frag_b(sB[cur], wn * 32 + 16, lane);
    acc0 = WMMA_BF16(a, b0, acc0);
    acc1 = WMMA_BF16(a, b1, acc1);

    wait_async();
    __syncthreads();
  }

  const int l16 = lane & 15, lhi = lane >> 4;
  const int row0 = m0 + wm * 16;
#pragma unroll
  for (int t = 0; t < 2; ++t) {
    const v8f& a = t ? acc1 : acc0;
    const int n = n0 + wn * 32 + t * 16 + l16;
#pragma unroll
    for (int r = 0; r < 8; ++r) {
      const int m = row0 + r + lhi * 8;
      h[(size_t)m * HDIM + n] = a[r] + bfv[n];
    }
  }
}

// ---------------------------------------------------------------------------
// Kernel 5: LayerNorm + ReLU + residual add.  One block per token.
// ---------------------------------------------------------------------------
__global__ __launch_bounds__(256) void ln_kernel(
    const float* __restrict__ h, const float* __restrict__ item,
    const float* __restrict__ g, const float* __restrict__ b,
    float* __restrict__ out) {
  __shared__ float red[256];
  const int tok = blockIdx.x;
  const int tid = threadIdx.x;
  const float* hr = h + (size_t)tok * HDIM;

  float v[4];
  float s = 0.f;
#pragma unroll
  for (int j = 0; j < 4; ++j) { v[j] = hr[tid + j * 256]; s += v[j]; }
  red[tid] = s; __syncthreads();
  for (int off = 128; off > 0; off >>= 1) {
    if (tid < off) red[tid] += red[tid + off];
    __syncthreads();
  }
  const float mean = red[0] * (1.f / HDIM);
  __syncthreads();

  float vs = 0.f;
#pragma unroll
  for (int j = 0; j < 4; ++j) { const float d = v[j] - mean; vs += d * d; }
  red[tid] = vs; __syncthreads();
  for (int off = 128; off > 0; off >>= 1) {
    if (tid < off) red[tid] += red[tid + off];
    __syncthreads();
  }
  const float rstd = rsqrtf(red[0] * (1.f / HDIM) + 1e-5f);

#pragma unroll
  for (int j = 0; j < 4; ++j) {
    const int n = tid + j * 256;
    float x = (v[j] - mean) * rstd * g[n] + b[n];
    x = fmaxf(x, 0.f);
    out[(size_t)tok * HDIM + n] = item[(size_t)tok * HDIM + n] + x;
  }
}

// ---------------------------------------------------------------------------
extern "C" void kernel_launch(void* const* d_in, const int* in_sizes, int n_in,
                              void* d_out, int out_size, void* d_ws,
                              size_t ws_size, hipStream_t stream) {
  const float* text    = (const float*)d_in[0];
  const float* img     = (const float*)d_in[1];
  const float* item    = (const float*)d_in[2];
  const float* eps_t   = (const float*)d_in[3];
  const float* eps_i   = (const float*)d_in[4];
  const float* W_mu_t  = (const float*)d_in[5];
  const float* b_mu_t  = (const float*)d_in[6];
  const float* W_sig_t = (const float*)d_in[7];
  const float* b_sig_t = (const float*)d_in[8];
  const float* W_mu_i  = (const float*)d_in[9];
  const float* b_mu_i  = (const float*)d_in[10];
  const float* W_sig_i = (const float*)d_in[11];
  const float* b_sig_i = (const float*)d_in[12];
  const float* W_gate  = (const float*)d_in[13];
  const float* b_gate  = (const float*)d_in[14];
  const float* W_texp  = (const float*)d_in[15];
  const float* b_texp  = (const float*)d_in[16];
  const float* W_iexp  = (const float*)d_in[17];
  const float* b_iexp  = (const float*)d_in[18];
  const float* W_fuse  = (const float*)d_in[19];
  const float* b_fuse  = (const float*)d_in[20];
  const float* ln_g    = (const float*)d_in[21];
  const float* ln_b    = (const float*)d_in[22];
  // d_in[23] = k (always 2; top-2 hardcoded in gate_kernel)

  char* ws = (char*)d_ws;
  size_t off = 0;
  auto bfbuf = [&](size_t elems) {
    __bf16* p = (__bf16*)(ws + off); off += elems * sizeof(__bf16); return p;
  };
  auto f32buf = [&](size_t elems) {
    float* p = (float*)(ws + off); off += elems * sizeof(float); return p;
  };
  const size_t NTOKH = (size_t)BLTOK * HDIM;   // 8M elems
  const size_t HH    = (size_t)HDIM * HDIM;    // 1M elems

  __bf16* xt  = bfbuf(NTOKH);
  __bf16* xi  = bfbuf(NTOKH);
  __bf16* wmt = bfbuf(HH);
  __bf16* wst = bfbuf(HH);
  __bf16* wmi = bfbuf(HH);
  __bf16* wsi = bfbuf(HH);
  __bf16* wte = bfbuf(NEXP * HH);
  __bf16* wie = bfbuf(NEXP * HH);
  __bf16* wfu = bfbuf(2 * HH);
  __bf16* z_t = bfbuf(NTOKH);
  __bf16* z_i = bfbuf(NTOKH);
  __bf16* o_t = bfbuf(NTOKH);
  __bf16* o_i = bfbuf(NTOKH);
  float*  g_t = f32buf((size_t)BLTOK * NEXP);
  float*  g_i = f32buf((size_t)BLTOK * NEXP);
  float*  hb  = f32buf(NTOKH);

  const dim3 blk(256);
  auto cvt = [&](const float* s, __bf16* d, size_t n) {
    cvt_bf16_kernel<<<dim3((unsigned)(n / 1024)), blk, 0, stream>>>(s, d);
  };
  cvt(text,   xt,  NTOKH);
  cvt(img,    xi,  NTOKH);
  cvt(W_mu_t, wmt, HH);
  cvt(W_sig_t,wst, HH);
  cvt(W_mu_i, wmi, HH);
  cvt(W_sig_i,wsi, HH);
  cvt(W_texp, wte, NEXP * HH);
  cvt(W_iexp, wie, NEXP * HH);
  cvt(W_fuse, wfu, 2 * HH);

  const dim3 gGemm(HDIM / BN, BLTOK / BM);  // (8, 256)

  proj_z_kernel<<<gGemm, blk, 0, stream>>>(xt, wmt, b_mu_t, wst, b_sig_t,
                                           eps_t, z_t);
  proj_z_kernel<<<gGemm, blk, 0, stream>>>(xi, wmi, b_mu_i, wsi, b_sig_i,
                                           eps_i, z_i);
  gate_kernel<<<dim3(BLTOK / 8), blk, 0, stream>>>(z_t, W_gate, b_gate, g_t);
  gate_kernel<<<dim3(BLTOK / 8), blk, 0, stream>>>(z_i, W_gate, b_gate, g_i);
  moe_kernel<<<gGemm, blk, 0, stream>>>(z_t, wte, b_texp, g_t, o_t);
  moe_kernel<<<gGemm, blk, 0, stream>>>(z_i, wie, b_iexp, g_i, o_i);
  fuse_kernel<<<gGemm, blk, 0, stream>>>(o_t, o_i, wfu, b_fuse, hb);
  ln_kernel<<<dim3(BLTOK), blk, 0, stream>>>(hb, item, ln_g, ln_b,
                                             (float*)d_out);
}
